// MultiDilatelocalAttention_43327630082583
// MI455X (gfx1250) — matrix operand order repaired and checked by reading, not compiled, not run on hardware
//
#include <hip/hip_runtime.h>
#include <hip/hip_bf16.h>
#include <math.h>

// ---------------------------------------------------------------------------
// Multi-Dilate Local Attention block for MI455X (gfx1250, wave32, WMMA).
// GEMMs: v_wmma_f32_16x16x32_bf16, double-buffered LDS filled by the async
// global->LDS engine (global_load_async_to_lds_b128 / ASYNCcnt, inline asm).
// ---------------------------------------------------------------------------

typedef __attribute__((ext_vector_type(8)))  float  v8f;
typedef __attribute__((ext_vector_type(16))) __bf16 v16bf;

union Frag {
    v16bf v;
    uint4 q[2];   // 2 x 16 bytes = 16 bf16
};

// Problem constants (from reference setup_inputs)
enum : int {
    Bb   = 8,
    Cc   = 256,
    Hh   = 64,
    Ww   = 64,
    HW   = 4096,    // Hh*Ww
    NTOK = 32768,   // Bb*HW
    HID  = 1024,
};

enum { GEMM_QKV = 0, GEMM_PROJ = 1, GEMM_FC1 = 2, GEMM_FC2 = 3 };

// GEMM LDS layout (dynamic LDS so raw byte offsets are well-defined):
//   [0,      8192)  Xs buffer 0   (128 x 32 bf16)
//   [8192,  16384)  Xs buffer 1
//   [16384, 20480)  Ws buffer 0   (64 x 32 bf16)
//   [20480, 24576)  Ws buffer 1
enum : unsigned {
    XS_BYTES  = 128 * 32 * 2,
    WS_BYTES  = 64 * 32 * 2,
    WS_BASE   = 2 * XS_BYTES,
    LDS_TOTAL = 2 * XS_BYTES + 2 * WS_BYTES,   // 24576 B
};

// Async global->LDS copy of 16 bytes per lane (Tensor-Access engine,
// tracked by ASYNCcnt; no VGPR round trip).
__device__ __forceinline__
void async_copy_b128(unsigned lds_byte_off, const void* gaddr)
{
    asm volatile("global_load_async_to_lds_b128 %0, %1, off"
                 :: "v"(lds_byte_off), "v"(gaddr)
                 : "memory");
}

__device__ __forceinline__ void wait_async_0()
{
    asm volatile("s_wait_asynccnt 0x0" ::: "memory");
}

// ---------------------------------------------------------------------------
// f32 [B,C,HW] (channels-first) -> bf16 [B*HW, C] (token-major)
// ---------------------------------------------------------------------------
__global__ __launch_bounds__(256)
void pack_tokens_kernel(const float* __restrict__ in, __bf16* __restrict__ out)
{
    size_t idx = (size_t)blockIdx.x * 256 + threadIdx.x;   // over B*C*HW
    int    hw  = (int)(idx & (HW - 1));
    size_t t   = idx >> 12;
    int    c   = (int)(t & (Cc - 1));
    size_t b   = t >> 8;
    out[((b << 12) + hw) * Cc + c] = (__bf16)in[idx];
}

// elementwise f32 -> bf16 (weights)
__global__ __launch_bounds__(256)
void convert_f32_bf16_kernel(const float* __restrict__ in,
                             __bf16* __restrict__ out, int n)
{
    int i = blockIdx.x * 256 + threadIdx.x;
    if (i < n) out[i] = (__bf16)in[i];
}

// ---------------------------------------------------------------------------
// Stage one K-slab (X: 128x32, Wt: 64x32) into LDS buffer `buf` via the
// async global->LDS engine. Per thread: 3 x 16-byte chunks.
// ---------------------------------------------------------------------------
__device__ __forceinline__
void stage_tiles(const __bf16* __restrict__ X, const __bf16* __restrict__ Wt,
                 unsigned lds_base, int buf,
                 int m0, int n0, int k0, int K, int t)
{
    const int r0 = t >> 2;              // 0..63
    const int cg = (t & 3) << 3;        // 0,8,16,24
    const unsigned xs_off = lds_base + (unsigned)buf * XS_BYTES;
    const unsigned ws_off = lds_base + WS_BASE + (unsigned)buf * WS_BYTES;

    async_copy_b128(xs_off + (unsigned)((r0 * 32 + cg) * 2),
                    &X[(size_t)(m0 + r0) * K + (k0 + cg)]);
    async_copy_b128(xs_off + (unsigned)(((r0 + 64) * 32 + cg) * 2),
                    &X[(size_t)(m0 + r0 + 64) * K + (k0 + cg)]);
    async_copy_b128(ws_off + (unsigned)((r0 * 32 + cg) * 2),
                    &Wt[(size_t)(n0 + r0) * K + (k0 + cg)]);
}

// ---------------------------------------------------------------------------
// Tiled WMMA GEMM:  Y[m,n] = sum_k X[m,k] * Wt[n,k]  (+bias, epilogue by MODE)
// X: [M,K] bf16 row-major, Wt: [N,K] bf16 row-major (i.e. Y = X * Wt^T).
// Block = 256 threads (8 waves). Macro-tile 128(M) x 64(N). K step 32.
// Double-buffered LDS (async fill); wave tile 32x32 = 2x2 WMMA accumulators.
// ---------------------------------------------------------------------------
template <int MODE>
__global__ __launch_bounds__(256, 2)
void wmma_gemm_kernel(const __bf16* __restrict__ X,
                      const __bf16* __restrict__ Wt,
                      const float*  __restrict__ bias,
                      void*         __restrict__ Yout,
                      const float*  __restrict__ resid,
                      int M, int N, int K)
{
    extern __shared__ char smem_raw[];
    const unsigned lds_base = __builtin_amdgcn_groupstaticsize();

    const int t    = threadIdx.x;
    const int m0   = blockIdx.x * 128;
    const int n0   = blockIdx.y * 64;
    const int wave = t >> 5;
    const int lane = t & 31;
    const int wm   = wave & 3;    // 0..3 over M
    const int wn   = wave >> 2;   // 0..1 over N
    const int lm   = lane & 15;
    const int lh   = lane >> 4;

    v8f acc[2][2] = {};

    const int nst = K >> 5;
    stage_tiles(X, Wt, lds_base, 0, m0, n0, 0, K, t);

    for (int s = 0; s < nst; ++s) {
        const int cur = s & 1;
        wait_async_0();      // my stage-`cur` copies have landed in LDS
        __syncthreads();     // everyone's copies landed; prior reads retired
        if (s + 1 < nst)
            stage_tiles(X, Wt, lds_base, cur ^ 1,
                        m0, n0, (s + 1) << 5, K, t);

        const __bf16* Xs = (const __bf16*)(smem_raw + cur * XS_BYTES);
        const __bf16* Ws = (const __bf16*)(smem_raw + WS_BASE + cur * WS_BYTES);

        // A fragment (16x32, MxK): lane lm holds row; lane-half lh selects
        // K chunks [lh*8, lh*8+8) and [lh*8+16, lh*8+24)  (ISA 7.12.2).
        Frag a[2], bfr[2];
        #pragma unroll
        for (int i = 0; i < 2; ++i) {
            int row = (wm * 32 + i * 16 + lm) * 32;
            int kb  = lh * 8;
            a[i].q[0] = *(const uint4*)&Xs[row + kb];
            a[i].q[1] = *(const uint4*)&Xs[row + kb + 16];
        }
        // B fragment (32x16, KxN): lane lm holds column n; lane-half lh
        // selects K chunks [lh*8, lh*8+8) and [lh*8+16, lh*8+24).
        #pragma unroll
        for (int j = 0; j < 2; ++j) {
            int col = (wn * 32 + j * 16 + lm) * 32;
            int kb  = lh * 8;
            bfr[j].q[0] = *(const uint4*)&Ws[col + kb];
            bfr[j].q[1] = *(const uint4*)&Ws[col + kb + 16];
        }

        #pragma unroll
        for (int i = 0; i < 2; ++i)
            #pragma unroll
            for (int j = 0; j < 2; ++j)
                acc[i][j] = __builtin_amdgcn_wmma_f32_16x16x32_bf16(
                    false, a[i].v, false, bfr[j].v,
                    (short)0, acc[i][j], false, false);
    }

    // Epilogue. C/D layout: lane lm = column n; VGPR r holds row r + lh*8.
    #pragma unroll
    for (int i = 0; i < 2; ++i) {
        #pragma unroll
        for (int j = 0; j < 2; ++j) {
            int   n  = n0 + wn * 32 + j * 16 + lm;
            float bv = (MODE == GEMM_QKV) ? 0.0f : bias[n];
            #pragma unroll
            for (int r = 0; r < 8; ++r) {
                int   m   = m0 + wm * 32 + i * 16 + lh * 8 + r;
                float val = acc[i][j][r] + bv;
                if (MODE == GEMM_QKV) {
                    // bf16, channels-first [B, C, HW] for the attention pass
                    ((__bf16*)Yout)[(((size_t)(m >> 12) * Cc + n) << 12) +
                                    (m & (HW - 1))] = (__bf16)val;
                } else if (MODE == GEMM_PROJ) {
                    ((float*)Yout)[(size_t)m * Cc + n] = val;
                } else if (MODE == GEMM_FC1) {
                    // exact GELU: 0.5*x*(1+erf(x/sqrt(2)))
                    float g = 0.5f * val *
                              (1.0f + erff(val * 0.7071067811865475f));
                    ((__bf16*)Yout)[(size_t)m * HID + n] = (__bf16)g;
                } else { // GEMM_FC2: residual add + final BCHW transpose
                    float o = resid[(size_t)m * Cc + n] + val;
                    ((float*)Yout)[(((size_t)(m >> 12) * Cc + n) << 12) +
                                   (m & (HW - 1))] = o;
                }
            }
        }
    }
}

// ---------------------------------------------------------------------------
// Multi-dilate local attention. One thread per (batch, head, pixel).
// Head gh in 0..7; dilation branch = gh>>1 (2 heads/branch), dil = branch+1.
// Channel slice for head gh is [gh*32, gh*32+32) in both q/k/v and the
// stacked output (branch-major stacking collapses to gh*32).
// Zero padding: OOB logit = 0 (counts in softmax denom), OOB v contributes 0.
// ---------------------------------------------------------------------------
__global__ __launch_bounds__(256)
void dilate_attn_kernel(const __bf16* __restrict__ q,
                        const __bf16* __restrict__ k,
                        const __bf16* __restrict__ v,
                        __bf16* __restrict__ xattn)   // [NTOK, C] bf16
{
    int tid = blockIdx.x * 256 + threadIdx.x;     // over B*8*HW
    int p   = tid & (HW - 1);
    int bg  = tid >> 12;
    int gh  = bg & 7;
    int b   = bg >> 3;
    int y   = p >> 6;
    int x   = p & 63;
    int dil = (gh >> 1) + 1;

    const size_t base = ((size_t)b * Cc + gh * 32) * HW;
    const float  scale = 0.17677669529663687f;    // 1/sqrt(32)

    float qv[32];
    #pragma unroll
    for (int c = 0; c < 32; ++c)
        qv[c] = (float)q[base + (size_t)c * HW + p];

    float logits[9];
    #pragma unroll
    for (int kk = 0; kk < 9; ++kk) {
        int dy = (kk / 3 - 1) * dil;
        int dx = (kk % 3 - 1) * dil;
        int ny = y + dy, nx = x + dx;
        float s = 0.0f;
        if ((unsigned)ny < (unsigned)Hh && (unsigned)nx < (unsigned)Ww) {
            int np = ny * Ww + nx;
            #pragma unroll
            for (int c = 0; c < 32; ++c)
                s += qv[c] * (float)k[base + (size_t)c * HW + np];
        }
        logits[kk] = s * scale;
    }

    float mx = logits[0];
    #pragma unroll
    for (int kk = 1; kk < 9; ++kk) mx = fmaxf(mx, logits[kk]);
    float wgt[9], wsum = 0.0f;
    #pragma unroll
    for (int kk = 0; kk < 9; ++kk) {
        wgt[kk] = __expf(logits[kk] - mx);
        wsum += wgt[kk];
    }
    float inv = 1.0f / wsum;

    float out[32] = {};
    #pragma unroll
    for (int kk = 0; kk < 9; ++kk) {
        int dy = (kk / 3 - 1) * dil;
        int dx = (kk % 3 - 1) * dil;
        int ny = y + dy, nx = x + dx;
        if ((unsigned)ny < (unsigned)Hh && (unsigned)nx < (unsigned)Ww) {
            int   np  = ny * Ww + nx;
            float wk_ = wgt[kk] * inv;
            #pragma unroll
            for (int c = 0; c < 32; ++c)
                out[c] += wk_ * (float)v[base + (size_t)c * HW + np];
        }
    }

    size_t m = (size_t)b * HW + p;
    #pragma unroll
    for (int c = 0; c < 32; ++c)
        xattn[m * Cc + gh * 32 + c] = (__bf16)out[c];
}

// ---------------------------------------------------------------------------
// LayerNorm over C=256, one wave32 per token, butterfly reductions.
// ---------------------------------------------------------------------------
__global__ __launch_bounds__(256)
void layernorm_kernel(const float* __restrict__ xin,
                      const float* __restrict__ w,
                      const float* __restrict__ bparm,
                      __bf16* __restrict__ yout)
{
    int tok  = (blockIdx.x * 256 + threadIdx.x) >> 5;
    int lane = threadIdx.x & 31;
    const float* row = xin + (size_t)tok * Cc;

    float vals[8], s = 0.0f;
    #pragma unroll
    for (int i = 0; i < 8; ++i) {
        vals[i] = row[lane * 8 + i];
        s += vals[i];
    }
    #pragma unroll
    for (int off = 16; off > 0; off >>= 1) s += __shfl_xor(s, off, 32);
    float mu = s * (1.0f / Cc);

    float vsum = 0.0f;
    #pragma unroll
    for (int i = 0; i < 8; ++i) {
        float d = vals[i] - mu;
        vsum += d * d;
    }
    #pragma unroll
    for (int off = 16; off > 0; off >>= 1) vsum += __shfl_xor(vsum, off, 32);
    float rstd = rsqrtf(vsum * (1.0f / Cc) + 1e-5f);

    #pragma unroll
    for (int i = 0; i < 8; ++i) {
        int c = lane * 8 + i;
        yout[(size_t)tok * Cc + c] =
            (__bf16)((vals[i] - mu) * rstd * w[c] + bparm[c]);
    }
}

// ---------------------------------------------------------------------------
// Host launcher
// ---------------------------------------------------------------------------
extern "C" void kernel_launch(void* const* d_in, const int* in_sizes, int n_in,
                              void* d_out, int out_size, void* d_ws, size_t ws_size,
                              hipStream_t stream)
{
    const float* sub   = (const float*)d_in[0];
    const float* ori   = (const float*)d_in[1];
    const float* wq    = (const float*)d_in[2];
    const float* wk    = (const float*)d_in[3];
    const float* wv    = (const float*)d_in[4];
    const float* projw = (const float*)d_in[5];
    const float* projb = (const float*)d_in[6];
    const float* lnw   = (const float*)d_in[7];
    const float* lnb   = (const float*)d_in[8];
    const float* fc1w  = (const float*)d_in[9];
    const float* fc1b  = (const float*)d_in[10];
    const float* fc2w  = (const float*)d_in[11];
    const float* fc2b  = (const float*)d_in[12];

    char*  ws  = (char*)d_ws;
    size_t off = 0;
    auto alloc = [&](size_t bytes) -> char* {
        char* p = ws + off;
        off += (bytes + 255) & ~(size_t)255;
        return p;
    };

    __bf16* sub_tok = (__bf16*)alloc((size_t)NTOK * Cc * 2);
    __bf16* ori_tok = (__bf16*)alloc((size_t)NTOK * Cc * 2);
    __bf16* wq_b    = (__bf16*)alloc((size_t)Cc * Cc * 2);
    __bf16* wk_b    = (__bf16*)alloc((size_t)Cc * Cc * 2);
    __bf16* wv_b    = (__bf16*)alloc((size_t)Cc * Cc * 2);
    __bf16* projw_b = (__bf16*)alloc((size_t)Cc * Cc * 2);
    __bf16* fc1w_b  = (__bf16*)alloc((size_t)HID * Cc * 2);
    __bf16* fc2w_b  = (__bf16*)alloc((size_t)Cc * HID * 2);
    __bf16* qbuf    = (__bf16*)alloc((size_t)NTOK * Cc * 2);   // [B,C,HW]
    __bf16* kbuf    = (__bf16*)alloc((size_t)NTOK * Cc * 2);
    __bf16* vbuf    = (__bf16*)alloc((size_t)NTOK * Cc * 2);
    __bf16* xattn   = (__bf16*)alloc((size_t)NTOK * Cc * 2);   // [N,C]
    float*  xproj   = (float*) alloc((size_t)NTOK * Cc * 4);   // [N,C]
    __bf16* yln     = (__bf16*)alloc((size_t)NTOK * Cc * 2);   // [N,C]
    __bf16* hbuf    = (__bf16*)alloc((size_t)NTOK * HID * 2);  // [N,HID]

    // 1. pack activations + convert weights to bf16
    pack_tokens_kernel<<<(Bb * Cc * HW) / 256, 256, 0, stream>>>(sub, sub_tok);
    pack_tokens_kernel<<<(Bb * Cc * HW) / 256, 256, 0, stream>>>(ori, ori_tok);
    convert_f32_bf16_kernel<<<(Cc * Cc + 255) / 256, 256, 0, stream>>>(wq, wq_b, Cc * Cc);
    convert_f32_bf16_kernel<<<(Cc * Cc + 255) / 256, 256, 0, stream>>>(wk, wk_b, Cc * Cc);
    convert_f32_bf16_kernel<<<(Cc * Cc + 255) / 256, 256, 0, stream>>>(wv, wv_b, Cc * Cc);
    convert_f32_bf16_kernel<<<(Cc * Cc + 255) / 256, 256, 0, stream>>>(projw, projw_b, Cc * Cc);
    convert_f32_bf16_kernel<<<(HID * Cc + 255) / 256, 256, 0, stream>>>(fc1w, fc1w_b, HID * Cc);
    convert_f32_bf16_kernel<<<(Cc * HID + 255) / 256, 256, 0, stream>>>(fc2w, fc2w_b, Cc * HID);

    // 2. Q/K/V projections (WMMA), output channels-first bf16
    dim3 gqkv(NTOK / 128, Cc / 64);
    wmma_gemm_kernel<GEMM_QKV><<<gqkv, 256, LDS_TOTAL, stream>>>(
        sub_tok, wq_b, nullptr, qbuf, nullptr, NTOK, Cc, Cc);
    wmma_gemm_kernel<GEMM_QKV><<<gqkv, 256, LDS_TOTAL, stream>>>(
        ori_tok, wk_b, nullptr, kbuf, nullptr, NTOK, Cc, Cc);
    wmma_gemm_kernel<GEMM_QKV><<<gqkv, 256, LDS_TOTAL, stream>>>(
        ori_tok, wv_b, nullptr, vbuf, nullptr, NTOK, Cc, Cc);

    // 3. multi-dilate local attention -> token-major bf16
    dilate_attn_kernel<<<(Bb * 8 * HW) / 256, 256, 0, stream>>>(
        qbuf, kbuf, vbuf, xattn);

    // 4. output projection (+bias) -> f32 xproj
    wmma_gemm_kernel<GEMM_PROJ><<<gqkv, 256, LDS_TOTAL, stream>>>(
        xattn, projw_b, projb, xproj, nullptr, NTOK, Cc, Cc);

    // 5. LayerNorm -> bf16
    layernorm_kernel<<<NTOK / 8, 256, 0, stream>>>(xproj, lnw, lnb, yln);

    // 6. fc1 (+bias, GELU) -> bf16 hidden
    dim3 gfc1(NTOK / 128, HID / 64);
    wmma_gemm_kernel<GEMM_FC1><<<gfc1, 256, LDS_TOTAL, stream>>>(
        yln, fc1w_b, fc1b, hbuf, nullptr, NTOK, HID, Cc);

    // 7. fc2 (+bias) + residual + final [B,C,H,W] transpose -> d_out (f32)
    wmma_gemm_kernel<GEMM_FC2><<<gqkv, 256, LDS_TOTAL, stream>>>(
        hbuf, fc2w_b, fc2b, d_out, xproj, NTOK, Cc, HID);
}
